// TokenFlow_1726576856370
// MI455X (gfx1250) — matrix operand order repaired and checked by reading, not compiled, not run on hardware
//
#include <hip/hip_runtime.h>

// ---------------------------------------------------------------------------
// TokenFlow attention on MI455X (gfx1250).
// bf16 WMMA (v_wmma_f32_16x16x32_bf16) + f32 accum: ~137 GFLOP vs ~150MB
// traffic -> matrix-core bound, so keep WMMAs fed:
//  * K tiles staged by the Tensor Data Mover (tensor_load_to_lds, TENSORcnt),
//    double-buffered so DMA of block i+1 overlaps compute of block i.
//  * V tiles prefetched to registers one block ahead, then transposed to LDS.
//  * dh=80 padded to 96 so the WMMA K-depth (32) divides evenly.
// ---------------------------------------------------------------------------

typedef __attribute__((ext_vector_type(8)))  __bf16 v8bf;
typedef __attribute__((ext_vector_type(16))) __bf16 v16bf;
typedef __attribute__((ext_vector_type(8)))  float  v8f;
typedef __attribute__((ext_vector_type(4)))  unsigned int v4u;
typedef __attribute__((ext_vector_type(8)))  int    v8i;
typedef __attribute__((ext_vector_type(4)))  int    v4i;
typedef unsigned short u16;
typedef unsigned int   u32;
typedef unsigned long long u64;

#define S_LEN   1024
#define D_DIM   640
#define HEADS   8
#define DH      80
#define DHP     96      // padded dh (multiple of 32)
#define NFRAMES 4

#if __has_builtin(__builtin_amdgcn_tensor_load_to_lds) && \
    __has_builtin(__builtin_amdgcn_s_wait_tensorcnt)
#define HAVE_TDM 1
#else
#define HAVE_TDM 0
#endif

__device__ __forceinline__ u16 f2bf(float f) {
    return __builtin_bit_cast(u16, (__bf16)f);   // native v_cvt on gfx1250
}

__device__ __forceinline__ v16bf cat8(v8bf lo, v8bf hi) {
    return __builtin_shufflevector(lo, hi, 0,1,2,3,4,5,6,7,8,9,10,11,12,13,14,15);
}

// A-operand (16x32, MxK), row-major source with row stride `rs` (elements).
// Lane L: row = L%16; VGPR0..3 hold K = 8*(L/16)+0..7, VGPR4..7 hold +16.
__device__ __forceinline__ v16bf load_frag_a(const u16* base, int rs, int lane) {
    const u16* p = base + (lane & 15) * rs + ((lane >> 4) << 3);
    return cat8(*(const v8bf*)(p), *(const v8bf*)(p + 16));
}

// B-operand (32x16, KxN) from a Bt[n][k] (transposed) tile with row stride `rs`.
// Lane L: col = L%16; VGPR0..3 hold K = 8*(L/16)+0..7, VGPR4..7 hold +16.
__device__ __forceinline__ v16bf load_frag_b(const u16* base, int rs, int lane) {
    const u16* p = base + (lane & 15) * rs + ((lane >> 4) << 3);
    return cat8(*(const v8bf*)(p), *(const v8bf*)(p + 16));
}

__device__ __forceinline__ v8f wmma_bf16(v16bf a, v16bf b, v8f c) {
    return __builtin_amdgcn_wmma_f32_16x16x32_bf16(false, a, false, b, (short)0, c,
                                                   false, false);
}

#if HAVE_TDM
// One 2-D TDM descriptor: tile 96 elems x 32 rows of 2-byte data, row stride 96.
// D# layout per CDNA5 ISA ch.8: group0 {count, lds_addr, global_addr, type=2},
// group1 {data_size=2B, tensor dims, tile dims, dim0 stride}.
__device__ __forceinline__ void tdm_load_k96x32(u32 lds_byte_addr, const u16* gptr) {
    u64 ga = (u64)(size_t)gptr;
    v4u g0 = { 1u,                                   // count=1 (valid), user mode
               lds_byte_addr,                        // lds_addr [63:32]
               (u32)(ga & 0xFFFFFFFFu),              // global_addr lo
               (u32)((ga >> 32) & 0x01FFFFFFu) | (2u << 30) };   // ga hi | type=2
    v8i g1 = { (int)(1u << 16),                      // data_size=1 -> 2 bytes
               (int)((u32)DHP << 16),                // tensor_dim0 = 96 (lo16)
               0,                                    // dim0 hi | tensor_dim1 lo16(=0)
               (int)(16u | ((u32)DHP << 16)),        // tensor_dim1 hi (1M) | tile_dim0=96
               32,                                   // tile_dim1 = 32 rows, tile_dim2=0
               DHP,                                  // tensor_dim0_stride = 96 (lo32)
               0, 0 };                               // stride hi / dim1 stride
    v4i z4 = {0, 0, 0, 0};
    v8i z8 = {0, 0, 0, 0, 0, 0, 0, 0};
    __builtin_amdgcn_tensor_load_to_lds(g0, g1, z4, z4, z8, 0);
}
#endif

// ---------------------------------------------------------------------------
// Kernel 1: QKV projection. C[12288x640] = x[12288x640] @ W[640x640].
// Block 256 thr (8 waves), tile 64(M) x 32(N), K-steps of 32.
// Output scattered into head-split padded layout Out[(b*8+h)*1024 + s][96].
// Q pre-scaled by dh^-0.5.
// ---------------------------------------------------------------------------
__global__ __launch_bounds__(256) void qkv_gemm(
    const float* __restrict__ x,
    const float* __restrict__ Wq, const float* __restrict__ Wk,
    const float* __restrict__ Wv,
    u16* __restrict__ Qp, u16* __restrict__ Kp, u16* __restrict__ Vp)
{
    const int z = blockIdx.z;
    const float* __restrict__ W = (z == 0) ? Wq : (z == 1) ? Wk : Wv;
    u16* __restrict__ Out       = (z == 0) ? Qp : (z == 1) ? Kp : Vp;
    const float oscale = (z == 0) ? rsqrtf((float)DH) : 1.0f;

    __shared__ __align__(32) u16 sA[64 * 32];
    __shared__ __align__(32) u16 sB[32 * 32];

    const int t = threadIdx.x, lane = t & 31, w = t >> 5;
    const int wm = w & 3, wn = w >> 2;             // waves: 4(M) x 2(N)
    const int m0 = blockIdx.y * 64;
    const int n0 = blockIdx.x * 32;

    v8f acc = {0.f, 0.f, 0.f, 0.f, 0.f, 0.f, 0.f, 0.f};

    for (int k0 = 0; k0 < D_DIM; k0 += 32) {
        if (k0 + 32 < D_DIM) {                     // prefetch next tiles into L2
            __builtin_prefetch(x + (size_t)(m0 + (t >> 2)) * D_DIM + k0 + 32, 0, 0);
            __builtin_prefetch(W + (size_t)(k0 + 32 + (t >> 3)) * D_DIM + n0, 0, 0);
        }
        // Stage A: 64x32 fp32 -> bf16 LDS (row-major)
        #pragma unroll
        for (int i = 0; i < 2; ++i) {
            int e4  = t + i * 256;                 // 512 float4
            int row = e4 >> 3;
            int c4  = e4 & 7;
            float4 f = *(const float4*)(x + (size_t)(m0 + row) * D_DIM + k0 + c4 * 4);
            u16* d = &sA[row * 32 + c4 * 4];
            d[0] = f2bf(f.x); d[1] = f2bf(f.y); d[2] = f2bf(f.z); d[3] = f2bf(f.w);
        }
        // Stage B transposed: sB[n][k] = W[k0+k][n0+n]
        #pragma unroll
        for (int i = 0; i < 4; ++i) {
            int e = t + i * 256;                   // 1024
            int kk = e >> 5, nn = e & 31;
            sB[nn * 32 + kk] = f2bf(W[(size_t)(k0 + kk) * D_DIM + n0 + nn]);
        }
        __syncthreads();
        v16bf a = load_frag_a(&sA[wm * 16 * 32], 32, lane);
        v16bf b = load_frag_b(&sB[wn * 16 * 32], 32, lane);
        acc = wmma_bf16(a, b, acc);
        __syncthreads();
    }

    // Epilogue: D-layout (row = v + 8*(lane/16), col = lane%16) -> head-split
    const int half = lane >> 4, nn = lane & 15;
    const int n = n0 + wn * 16 + nn;
    const int h = n / DH, d = n % DH;
    #pragma unroll
    for (int v = 0; v < 8; ++v) {
        int m = m0 + wm * 16 + v + 8 * half;
        int b = m >> 10, s = m & 1023;
        Out[((size_t)(b * HEADS + h) * S_LEN + s) * DHP + d] = f2bf(acc[v] * oscale);
    }
}

// ---------------------------------------------------------------------------
// Kernel 2: flash attention. Block = 128 thr (4 waves); each wave owns 16
// queries (block = 64 queries of one (b,h)). Keys processed in blocks of 32:
//   sK (double buffered) filled by the TDM one block ahead,
//   sV filled from registers prefetched one block ahead (transposed store),
//   online softmax via 16-lane shfl_xor reductions on the WMMA D-layout.
// ---------------------------------------------------------------------------
__global__ __launch_bounds__(128) void attn_kernel(
    const u16* __restrict__ Qp, const u16* __restrict__ Kp,
    const u16* __restrict__ Vp, u16* __restrict__ Ob)
{
    __shared__ __align__(32) u16 sK[2][32 * DHP];  // 2 x (32 keys x 96)  (12 KB)
    __shared__ __align__(32) u16 sV[DHP * 48];     // 96 d x 32 keys, stride 48
    __shared__ __align__(32) u16 sP[4][16 * 32];   // per-wave P tile

    const int t = threadIdx.x, lane = t & 31, w = t >> 5;
    const int qc = blockIdx.x;                     // 0..15
    const int h  = blockIdx.y;                     // 0..7
    const int b  = blockIdx.z;                     // 0..11
    const int g  = b >> 2, f = b & 3;

    const int q0 = qc * 64 + w * 16;
    const size_t qbase = ((size_t)(b * HEADS + h) * S_LEN + q0) * DHP;

    // Q A-fragments for the 3 K-steps of dh(96) -- loaded once, kept in VGPRs.
    v16bf qa[3];
    {
        const int row = lane & 15, h8 = (lane >> 4) << 3;
        #pragma unroll
        for (int ks = 0; ks < 3; ++ks) {
            const u16* p = Qp + qbase + (size_t)row * DHP + ks * 32 + h8;
            qa[ks] = cat8(*(const v8bf*)p, *(const v8bf*)(p + 16));
        }
    }

    float mrow[8], lrow[8];
    v8f o[6];
    #pragma unroll
    for (int v = 0; v < 8; ++v) { mrow[v] = -3.0e38f; lrow[v] = 0.f; }
    #pragma unroll
    for (int n = 0; n < 6; ++n) o[n] = (v8f){0.f,0.f,0.f,0.f,0.f,0.f,0.f,0.f};

    const int f_begin = (g == 0) ? f : 0;
    const int nb = ((g == 0) ? 1 : NFRAMES) * (S_LEN / 32);

    // idx -> (frame, key block) base pointers
    auto kbase = [&](int idx) -> const u16* {
        int ffi = f_begin + (idx >> 5);
        int kb  = (idx & 31) * 32;
        return Kp + ((size_t)((g * NFRAMES + ffi) * HEADS + h) * S_LEN + kb) * DHP;
    };
    auto vbase = [&](int idx) -> const u16* {
        int ffi = f_begin + (idx >> 5);
        int kb  = (idx & 31) * 32;
        return Vp + ((size_t)((g * NFRAMES + ffi) * HEADS + h) * S_LEN + kb) * DHP;
    };

    // V register prefetch: 6 x uint2 per thread = 24 bf16 (4 per chunk, row-local)
    uint2 vreg[6];
    auto preloadV = [&](const u16* vb) {
        #pragma unroll
        for (int i = 0; i < 6; ++i) {
            int c = t + i * 128;                   // 768 chunks of 4 elems
            int key = c / 24, d0 = (c % 24) * 4;
            vreg[i] = *(const uint2*)(vb + (size_t)key * DHP + d0);
        }
    };
    auto storeV = [&]() {
        #pragma unroll
        for (int i = 0; i < 6; ++i) {
            int c = t + i * 128;
            int key = c / 24, d0 = (c % 24) * 4;
            u32 lo = vreg[i].x, hi = vreg[i].y;
            sV[(d0 + 0) * 48 + key] = (u16)(lo & 0xFFFF);
            sV[(d0 + 1) * 48 + key] = (u16)(lo >> 16);
            sV[(d0 + 2) * 48 + key] = (u16)(hi & 0xFFFF);
            sV[(d0 + 3) * 48 + key] = (u16)(hi >> 16);
        }
    };

    preloadV(vbase(0));
#if HAVE_TDM
    if (w == 0) tdm_load_k96x32((u32)(size_t)(void*)&sK[0][0], kbase(0));
#endif

    for (int idx = 0; idx < nb; ++idx) {
        const int cur = idx & 1;
#if HAVE_TDM
        if (w == 0) __builtin_amdgcn_s_wait_tensorcnt(0);   // K[idx] in LDS
#endif
        __syncthreads();            // all waves past previous compute; sK/sV free

        storeV();                   // V[idx]: registers -> transposed LDS
#if !HAVE_TDM
        {   // fallback: synchronous K copy
            const uint4* src = (const uint4*)kbase(idx);
            uint4* dst = (uint4*)&sK[cur][0];
            #pragma unroll
            for (int i = 0; i < 3; ++i) dst[t + i * 128] = src[t + i * 128];
        }
#endif
        if (idx + 1 < nb) {
            preloadV(vbase(idx + 1));
#if HAVE_TDM
            if (w == 0)
                tdm_load_k96x32((u32)(size_t)(void*)&sK[cur ^ 1][0], kbase(idx + 1));
#endif
        }
        __syncthreads();            // sK[cur] + sV ready for everyone

        // S = Q·K^T for this 32-key block: two 16x16 subtiles
        v8f s0 = (v8f){0.f,0.f,0.f,0.f,0.f,0.f,0.f,0.f};
        v8f s1 = s0;
        #pragma unroll
        for (int ks = 0; ks < 3; ++ks) {
            s0 = wmma_bf16(qa[ks], load_frag_b(&sK[cur][ 0 * DHP + ks * 32], DHP, lane), s0);
            s1 = wmma_bf16(qa[ks], load_frag_b(&sK[cur][16 * DHP + ks * 32], DHP, lane), s1);
        }

        // Online softmax (scale folded into Q). Row = v + 8*(lane/16); the 16
        // N-values of a row live in one 16-lane group -> shfl_xor reductions.
        u16* pw = &sP[w][0];
        const int col = lane & 15, half = lane >> 4;
        #pragma unroll
        for (int v = 0; v < 8; ++v) {
            float m = fmaxf(s0[v], s1[v]);
            m = fmaxf(m, __shfl_xor(m, 1, 32));
            m = fmaxf(m, __shfl_xor(m, 2, 32));
            m = fmaxf(m, __shfl_xor(m, 4, 32));
            m = fmaxf(m, __shfl_xor(m, 8, 32));
            float mn = fmaxf(mrow[v], m);
            float al = __expf(mrow[v] - mn);
            float p0 = __expf(s0[v] - mn);
            float p1 = __expf(s1[v] - mn);
            float rs = p0 + p1;
            rs += __shfl_xor(rs, 1, 32);
            rs += __shfl_xor(rs, 2, 32);
            rs += __shfl_xor(rs, 4, 32);
            rs += __shfl_xor(rs, 8, 32);
            lrow[v] = lrow[v] * al + rs;
            mrow[v] = mn;
            #pragma unroll
            for (int n = 0; n < 6; ++n) o[n][v] = o[n][v] * al;
            int prow = v + 8 * half;
            pw[prow * 32 + col]      = f2bf(p0);
            pw[prow * 32 + 16 + col] = f2bf(p1);
        }
        __syncthreads();            // P visible to this wave's other lanes

        // O += P·V : A = P (16q x 32k), B = Vt rows (6 d-tiles of 16)
        v16bf pa = load_frag_a(pw, 32, lane);
        #pragma unroll
        for (int n = 0; n < 6; ++n)
            o[n] = wmma_bf16(pa, load_frag_b(&sV[(n * 16) * 48], 48, lane), o[n]);
    }

    // Normalize and store (batch_to_head_dim layout: Ob[b][s][h*80 + d], bf16)
    const int col = lane & 15, half = lane >> 4;
    #pragma unroll
    for (int v = 0; v < 8; ++v) {
        float inv = 1.0f / lrow[v];
        int q = q0 + v + 8 * half;
        size_t rowoff = ((size_t)b * S_LEN + q) * D_DIM + h * DH;
        #pragma unroll
        for (int n = 0; n < 6; ++n) {
            int d = n * 16 + col;
            if (d < DH) Ob[rowoff + d] = f2bf(o[n][v] * inv);
        }
    }
}

// ---------------------------------------------------------------------------
// Kernel 3: out = Ob(bf16) @ Wo + bo, fp32 output. Same tiling as kernel 1.
// ---------------------------------------------------------------------------
__global__ __launch_bounds__(256) void out_gemm(
    const u16* __restrict__ Ob, const float* __restrict__ Wo,
    const float* __restrict__ bo, float* __restrict__ out)
{
    __shared__ __align__(32) u16 sA[64 * 32];
    __shared__ __align__(32) u16 sB[32 * 32];

    const int t = threadIdx.x, lane = t & 31, w = t >> 5;
    const int wm = w & 3, wn = w >> 2;
    const int m0 = blockIdx.y * 64;
    const int n0 = blockIdx.x * 32;

    v8f acc = {0.f, 0.f, 0.f, 0.f, 0.f, 0.f, 0.f, 0.f};

    for (int k0 = 0; k0 < D_DIM; k0 += 32) {
        if (k0 + 32 < D_DIM) {
            __builtin_prefetch(Ob + (size_t)(m0 + (t >> 2)) * D_DIM + k0 + 32, 0, 0);
            __builtin_prefetch(Wo + (size_t)(k0 + 32 + (t >> 3)) * D_DIM + n0, 0, 0);
        }
        {   // Stage A: bf16 straight copy, one 16B chunk per thread
            int row = t >> 2, c = t & 3;
            *(uint4*)&sA[row * 32 + c * 8] =
                *(const uint4*)(Ob + (size_t)(m0 + row) * D_DIM + k0 + c * 8);
        }
        // Stage B transposed from fp32 Wo
        #pragma unroll
        for (int i = 0; i < 4; ++i) {
            int e = t + i * 256;
            int kk = e >> 5, nn = e & 31;
            sB[nn * 32 + kk] = f2bf(Wo[(size_t)(k0 + kk) * D_DIM + n0 + nn]);
        }
        __syncthreads();
        v16bf a = load_frag_a(&sA[wm * 16 * 32], 32, lane);
        v16bf b = load_frag_b(&sB[wn * 16 * 32], 32, lane);
        acc = wmma_bf16(a, b, acc);
        __syncthreads();
    }

    const int half = lane >> 4, nn = lane & 15;
    const int n = n0 + wn * 16 + nn;
    const float bias = bo[n];
    #pragma unroll
    for (int v = 0; v < 8; ++v) {
        int m = m0 + wm * 16 + v + 8 * half;
        out[(size_t)m * D_DIM + n] = acc[v] + bias;
    }
}

// ---------------------------------------------------------------------------
extern "C" void kernel_launch(void* const* d_in, const int* in_sizes, int n_in,
                              void* d_out, int out_size, void* d_ws, size_t ws_size,
                              hipStream_t stream) {
    const float* x  = (const float*)d_in[0];
    const float* Wq = (const float*)d_in[1];
    const float* Wk = (const float*)d_in[2];
    const float* Wv = (const float*)d_in[3];
    const float* Wo = (const float*)d_in[4];
    const float* bo = (const float*)d_in[5];
    float* out = (float*)d_out;

    char* ws = (char*)d_ws;
    const size_t QKV_BYTES = (size_t)12 * HEADS * S_LEN * DHP * sizeof(u16); // 18.9MB
    u16* Qp = (u16*)(ws);
    u16* Kp = (u16*)(ws + QKV_BYTES);
    u16* Vp = (u16*)(ws + 2 * QKV_BYTES);
    u16* Ob = (u16*)(ws + 3 * QKV_BYTES);          // 12288*640 bf16 (15.7MB)

    // Zero Q/K/V so the dh padding columns [80,96) stay zero.
    (void)hipMemsetAsync(ws, 0, 3 * QKV_BYTES, stream);

    qkv_gemm<<<dim3(D_DIM / 32, (12 * S_LEN) / 64, 3), 256, 0, stream>>>(
        x, Wq, Wk, Wv, Qp, Kp, Vp);
    attn_kernel<<<dim3(S_LEN / 64, HEADS, 12), 128, 0, stream>>>(Qp, Kp, Vp, Ob);
    out_gemm<<<dim3(D_DIM / 32, (12 * S_LEN) / 64, 1), 256, 0, stream>>>(
        Ob, Wo, bo, out);
}
